// NeuralAstar_9740985827597
// MI455X (gfx1250) — matrix operand order repaired and checked by reading, not compiled, not run on hardware
//
#include <hip/hip_runtime.h>
#include <hip/hip_bf16.h>

typedef _Float16 half_t;
typedef __attribute__((ext_vector_type(16))) _Float16 v16h;
typedef __attribute__((ext_vector_type(8)))  _Float16 v8h;
typedef __attribute__((ext_vector_type(8)))  float    v8f;

#define BATCH  16
#define HH     64
#define WW     64
#define HWTOT  4096
#define TSTEPS 1024   // 0.25 * 64 * 64

// ---------------------------------------------------------------------------
// Pack the 2-channel input (map, start+goal) into f16 NHWC.
// ---------------------------------------------------------------------------
__global__ void prep_input_k(const float* __restrict__ mapd, const float* __restrict__ st,
                             const float* __restrict__ gl, half_t* __restrict__ act) {
  int i = blockIdx.x * blockDim.x + threadIdx.x;
  if (i < BATCH * HWTOT) {
    act[2 * i + 0] = (half_t)mapd[i];
    act[2 * i + 1] = (half_t)(st[i] + gl[i]);
  }
}

// ---------------------------------------------------------------------------
// Pack conv weights [3][3][CIN][COUT] f32 -> wT[COUTP][KPAD] f16,
// K = (ky*3+kx)*CIN + ci, zero padded so WMMA B fragments are plain
// 16-byte (v8h) loads with no guards.
// ---------------------------------------------------------------------------
__global__ void pack_weights_k(const float* __restrict__ w, half_t* __restrict__ dst,
                               int CIN, int COUT, int KPAD, int COUTP) {
  int KT = 9 * CIN;
  int total = COUTP * KPAD;
  for (int i = blockIdx.x * blockDim.x + threadIdx.x; i < total;
       i += gridDim.x * blockDim.x) {
    int n = i / KPAD, K = i - n * KPAD;
    float v = 0.f;
    if (n < COUT && K < KT) {
      int tap = K / CIN, ci = K - tap * CIN;
      v = w[(tap * CIN + ci) * COUT + n];
    }
    dst[i] = (half_t)v;
  }
}

// ---------------------------------------------------------------------------
// Implicit-GEMM 3x3 SAME conv via v_wmma_f32_16x16x32_f16.
// One workgroup = 16 output pixels (one row segment) x all COUT channels.
// Wave w computes the 16-pixel x 16-channel tile [n0 = 16*w].
// A tile: im2row in LDS, srow[16][KPAD] f16 (b128 global->b128 ds staging
// for CIN>=8). B: packed wT rows (two 16B loads per lane).
// Epilogue: bias + BN(inference) + ReLU (or sigmoid on final layer).
// ---------------------------------------------------------------------------
template <int CIN, int COUT, bool FINAL>
__global__ void conv_wmma_k(const half_t* __restrict__ in, const half_t* __restrict__ wT,
                            const float* __restrict__ bias, const float* __restrict__ scale,
                            const float* __restrict__ offs, half_t* __restrict__ out_h,
                            float* __restrict__ out_f) {
  constexpr int KT    = 9 * CIN;
  constexpr int KPAD  = ((KT + 31) / 32) * 32;
  constexpr int COUTP = ((COUT + 15) / 16) * 16;
  constexpr int NW    = COUTP / 16;

  extern __shared__ char smem[];
  half_t* srow = (half_t*)smem;  // [16][KPAD]

  const int x0 = blockIdx.x * 16, y = blockIdx.y, b = blockIdx.z;
  const int tid = threadIdx.x;

  // ---- im2row staging (all waves cooperate) ----
  if constexpr (CIN >= 8) {
    // KPAD == KT here (9*CIN is a multiple of 32 for CIN in {32,64,128,256}).
    // 8-half chunks never cross a (ky,kx) tap: b128 global load -> b128 ds store.
    constexpr int CPK = KT / 8;           // chunks per pixel row
    for (int i = tid; i < 16 * CPK; i += blockDim.x) {
      int m = i / CPK, K8 = (i - m * CPK) * 8;
      int tap = K8 / CIN, ci = K8 - tap * CIN;
      int ky = tap / 3, kx = tap - ky * 3;
      int yi = y + ky - 1, xi = x0 + m + kx - 1;
      v8h v;
#pragma unroll
      for (int j = 0; j < 8; ++j) v[j] = (half_t)0.f;
      if ((unsigned)yi < (unsigned)HH && (unsigned)xi < (unsigned)WW)
        v = *(const v8h*)(in + ((b * HH + yi) * WW + xi) * CIN + ci);
      *(v8h*)(srow + m * KPAD + K8) = v;
    }
  } else {
    for (int i = tid; i < 16 * KPAD; i += blockDim.x) {
      int m = i / KPAD, K = i - m * KPAD;
      half_t v = (half_t)0.f;
      if (K < KT) {
        int tap = K / CIN, ci = K - tap * CIN;
        int ky = tap / 3, kx = tap - ky * 3;
        int yi = y + ky - 1, xi = x0 + m + kx - 1;
        if ((unsigned)yi < (unsigned)HH && (unsigned)xi < (unsigned)WW)
          v = in[((b * HH + yi) * WW + xi) * CIN + ci];
      }
      srow[i] = v;
    }
  }
  __syncthreads();

  const int wave = tid >> 5;
  if (wave < NW) {
    const int lane = tid & 31, hi = lane >> 4, l16 = lane & 15;
    const int n = wave * 16 + l16;

    v8f acc;
#pragma unroll
    for (int r = 0; r < 8; ++r) acc[r] = 0.f;

    const half_t* arow = srow + l16 * KPAD;   // A: row M = l16 (pixel)
    const half_t* wrow = wT + n * KPAD;       // B: column N = n

    for (int kk = 0; kk < KPAD; kk += 32) {
      if (kk + 32 < KPAD) __builtin_prefetch(wrow + kk + 64, 0, 0);
      // A fragment (16x32 f16): lane holds K chunks [kk+8*hi, +8) and [kk+16+8*hi, +8)
      v8h a0 = *(const v8h*)(arow + kk + 8 * hi);
      v8h a1 = *(const v8h*)(arow + kk + 16 + 8 * hi);
      v16h a;
#pragma unroll
      for (int j = 0; j < 8; ++j) { a[j] = a0[j]; a[j + 8] = a1[j]; }
      // B fragment (32x16 f16): lane holds K chunks [kk+8*hi, +8) and [kk+16+8*hi, +8) for column n
      v8h b0 = *(const v8h*)(wrow + kk + 8 * hi);
      v8h b1 = *(const v8h*)(wrow + kk + 16 + 8 * hi);
      v16h bf;
#pragma unroll
      for (int j = 0; j < 8; ++j) { bf[j] = b0[j]; bf[j + 8] = b1[j]; }
      acc = __builtin_amdgcn_wmma_f32_16x16x32_f16(false, a, false, bf,
                                                   (short)0, acc, false, false);
    }

    const float inv = 0.999995000037f;  // 1/sqrt(1 + 1e-5)  (BN inference)
    if (n < COUT) {
#pragma unroll
      for (int r = 0; r < 8; ++r) {
        int x = x0 + r + 8 * hi;            // C/D layout: VGPR r, half-wave hi -> M = r + 8*hi
        float val = acc[r] + bias[n];
        val = scale[n] * (val * inv) + offs[n];
        int pix = (b * HH + y) * WW + x;
        if (FINAL) {
          if (n == 0) out_f[pix] = 1.f / (1.f + __expf(-val));
        } else {
          out_h[pix * COUT + n] = (half_t)fmaxf(val, 0.f);
        }
      }
    }
  }
}

// ---------------------------------------------------------------------------
// Differentiable A* forward: one workgroup per batch map, all state in LDS.
// val[c] = exp(-(0.5*g+0.5*hmap)/8)*open is materialized in LDS and updated
// incrementally (only the 3x3 patch around the selection changes per step),
// so the per-step argmax is a pure b128 LDS scan + wave32 shuffle reduction.
// 1024 strictly sequential steps; backtrack at the end from parents[] in LDS.
// ---------------------------------------------------------------------------
__global__ __launch_bounds__(256) void astar_k(const float* __restrict__ cst,
                                               const float* __restrict__ obst,
                                               const float* __restrict__ st,
                                               const float* __restrict__ gl,
                                               float* __restrict__ hist_out,
                                               float* __restrict__ path_out) {
  extern __shared__ char smem[];
  float* hmap  = (float*)smem;
  float* g     = hmap + HWTOT;
  float* open  = g + HWTOT;
  float* hist  = open + HWTOT;
  float* val   = hist + HWTOT;
  int*   par   = (int*)(val + HWTOT);
  float* svals = (float*)(par + HWTOT);
  int*   sidx  = (int*)(svals + 8);
  int*   s_misc = sidx + 8;            // [0]=goal idx, [1]=selected idx
  float* s_v    = (float*)(s_misc + 2);

  const int b = blockIdx.x, tid = threadIdx.x;
  const int base = b * HWTOT;
  const float invs = 0.125f;  // 1/sqrt(W) = 1/8

  if (tid == 0) s_misc[0] = 0;
  __syncthreads();
  for (int c = tid; c < HWTOT; c += 256)
    if (gl[base + c] > 0.5f) s_misc[0] = c;
  __syncthreads();
  const int gi = s_misc[0];
  const float gy = (float)(gi >> 6), gx = (float)(gi & 63);

  // hmap = chebyshev + 0.001*euclid + cost ; init g/open/hist/val/parents
  for (int c = tid; c < HWTOT; c += 256) {
    float ry = (float)(c >> 6), rx = (float)(c & 63);
    float dy = fabsf(ry - gy), dx = fabsf(rx - gx);
    float cheb = fminf(dy, dx) + fabsf(dy - dx);
    float euc = sqrtf(dy * dy + dx * dx);
    float hm = cheb + 0.001f * euc + cst[base + c];
    float op = st[base + c];
    hmap[c] = hm;
    g[c] = 0.f;
    hist[c] = 0.f;
    open[c] = op;
    val[c] = __expf(-(0.5f * hm) * invs) * op;
    par[c] = gi;
  }
  __syncthreads();

  const int lane = tid & 31, wave = tid >> 5;
  const int cbase0 = tid << 4;   // 16 contiguous cells per thread

  for (int t = 0; t < TSTEPS; ++t) {
    // ---- hard argmax of val[], first-index ties (matches jnp.argmax) ----
    float bv = -1.f; int bi = 0;
#pragma unroll
    for (int q = 0; q < 4; ++q) {
      float4 v4 = *(const float4*)(val + cbase0 + (q << 2));
      int cb = cbase0 + (q << 2);
      if (v4.x > bv) { bv = v4.x; bi = cb; }
      if (v4.y > bv) { bv = v4.y; bi = cb + 1; }
      if (v4.z > bv) { bv = v4.z; bi = cb + 2; }
      if (v4.w > bv) { bv = v4.w; bi = cb + 3; }
    }
#pragma unroll
    for (int off = 16; off > 0; off >>= 1) {
      float ov = __shfl_down(bv, off);
      int   oi = __shfl_down(bi, off);
      if (ov > bv || (ov == bv && oi < bi)) { bv = ov; bi = oi; }
    }
    if (lane == 0) { svals[wave] = bv; sidx[wave] = bi; }
    __syncthreads();

    if (tid == 0) {
      float fb = svals[0]; int fi = sidx[0];
      for (int w2 = 1; w2 < 8; ++w2)
        if (svals[w2] > fb || (svals[w2] == fb && sidx[w2] < fi)) { fb = svals[w2]; fi = sidx[w2]; }
      s_misc[1] = fi;
      s_v[0] = g[fi] + cst[base + fi];
      hist[fi] = 1.f;                                       // clip(hist + sel)
      if (fi != gi) open[fi] = fmaxf(open[fi] - 1.f, 0.f);  // clip(open - unsolved*sel)
    }
    __syncthreads();

    const int sel = s_misc[1];
    const int sy = sel >> 6, sx = sel & 63;

    // ---- 8-neighbor relax (center excluded); neighbor cells are disjoint ----
    if (tid < 8) {
      int k = tid + (tid >= 4);         // skip center of 3x3
      int ny = sy + k / 3 - 1, nx = sx + (k % 3) - 1;
      if ((unsigned)ny < (unsigned)HH && (unsigned)nx < (unsigned)WW) {
        int nc = (ny << 6) | nx;
        if (obst[base + nc] > 0.f) {
          float v = s_v[0];
          float op = open[nc];
          bool take = (op > 0.f) ? (g[nc] > v) : (hist[nc] == 0.f);
          if (take) { g[nc] = v; open[nc] = 1.f; par[nc] = sel; }
        }
      }
    }
    __syncthreads();

    // ---- incremental val refresh over the 3x3 patch (center = sel) ----
    if (tid < 9) {
      int ny = sy + tid / 3 - 1, nx = sx + (tid % 3) - 1;
      if ((unsigned)ny < (unsigned)HH && (unsigned)nx < (unsigned)WW) {
        int c = (ny << 6) | nx;
        float f = 0.5f * g[c] + 0.5f * hmap[c];
        val[c] = __expf(-f * invs) * open[c];
      }
    }
    __syncthreads();
  }

  __builtin_amdgcn_s_wait_tensorcnt(0);  // CDNA5 split-counter wait (no-op here)

  // ---- outputs: histories, then path via backtrack ----
  for (int c = tid; c < HWTOT; c += 256) {
    hist_out[base + c] = hist[c];
    path_out[base + c] = (c == gi) ? 1.f : 0.f;
  }
  __syncthreads();
  if (tid == 0) {
    int loc = par[gi];
    for (int i = 0; i < TSTEPS; ++i) {
      path_out[base + loc] = 1.f;
      loc = par[loc];
    }
  }
}

// ---------------------------------------------------------------------------
extern "C" void kernel_launch(void* const* d_in, const int* in_sizes, int n_in,
                              void* d_out, int out_size, void* d_ws, size_t ws_size,
                              hipStream_t stream) {
  (void)in_sizes; (void)n_in; (void)out_size; (void)ws_size;

  const float* mapd = (const float*)d_in[0];
  const float* st   = (const float*)d_in[1];
  const float* gl   = (const float*)d_in[2];
  const float *W[5], *Bp[5], *Sp[5], *Op[5];
  for (int i = 0; i < 5; ++i) {
    W[i]  = (const float*)d_in[3 + 4 * i];
    Bp[i] = (const float*)d_in[4 + 4 * i];
    Sp[i] = (const float*)d_in[5 + 4 * i];
    Op[i] = (const float*)d_in[6 + 4 * i];
  }

  // workspace carve-up
  const size_t ACT_HALVES = (size_t)BATCH * HWTOT * 256;   // 16.7M halves = 32MB
  half_t* actA = (half_t*)d_ws;
  half_t* actB = actA + ACT_HALVES;
  half_t* wp   = actB + ACT_HALVES;
  // packed weight offsets (halves): [COUTP][KPAD]
  const size_t wo0 = 0;        // 32*32
  const size_t wo1 = 1024;     // 64*288
  const size_t wo2 = 19456;    // 128*576
  const size_t wo3 = 93184;    // 256*1152
  const size_t wo4 = 388096;   // 16*2304 ; end 424960
  float* cost = (float*)(wp + 425984);

  float* hist_out = (float*)d_out;
  float* path_out = hist_out + (size_t)BATCH * HWTOT;

  const int smem0 = 16 * 32 * 2;
  const int smem1 = 16 * 288 * 2;
  const int smem2 = 16 * 576 * 2;
  const int smem3 = 16 * 1152 * 2;
  const int smem4 = 16 * 2304 * 2;
  const int smemA = 6 * HWTOT * 4 + 128;

  (void)hipFuncSetAttribute((const void*)conv_wmma_k<64, 128, false>,
                            hipFuncAttributeMaxDynamicSharedMemorySize, smem2);
  (void)hipFuncSetAttribute((const void*)conv_wmma_k<128, 256, false>,
                            hipFuncAttributeMaxDynamicSharedMemorySize, smem3);
  (void)hipFuncSetAttribute((const void*)conv_wmma_k<256, 1, true>,
                            hipFuncAttributeMaxDynamicSharedMemorySize, smem4);
  (void)hipFuncSetAttribute((const void*)astar_k,
                            hipFuncAttributeMaxDynamicSharedMemorySize, smemA);

  prep_input_k<<<(BATCH * HWTOT + 255) / 256, 256, 0, stream>>>(mapd, st, gl, actA);

  pack_weights_k<<<256, 256, 0, stream>>>(W[0], wp + wo0, 2,   32,  32,   32);
  pack_weights_k<<<256, 256, 0, stream>>>(W[1], wp + wo1, 32,  64,  288,  64);
  pack_weights_k<<<256, 256, 0, stream>>>(W[2], wp + wo2, 64,  128, 576,  128);
  pack_weights_k<<<256, 256, 0, stream>>>(W[3], wp + wo3, 128, 256, 1152, 256);
  pack_weights_k<<<256, 256, 0, stream>>>(W[4], wp + wo4, 256, 1,   2304, 16);

  dim3 cgrid(WW / 16, HH, BATCH);
  conv_wmma_k<2, 32, false><<<cgrid, 64, smem0, stream>>>(actA, wp + wo0, Bp[0], Sp[0], Op[0], actB, nullptr);
  conv_wmma_k<32, 64, false><<<cgrid, 128, smem1, stream>>>(actB, wp + wo1, Bp[1], Sp[1], Op[1], actA, nullptr);
  conv_wmma_k<64, 128, false><<<cgrid, 256, smem2, stream>>>(actA, wp + wo2, Bp[2], Sp[2], Op[2], actB, nullptr);
  conv_wmma_k<128, 256, false><<<cgrid, 512, smem3, stream>>>(actB, wp + wo3, Bp[3], Sp[3], Op[3], actA, nullptr);
  conv_wmma_k<256, 1, true><<<cgrid, 128, smem4, stream>>>(actA, wp + wo4, Bp[4], Sp[4], Op[4], nullptr, cost);

  astar_k<<<BATCH, 256, smemA, stream>>>(cost, mapd, st, gl, hist_out, path_out);
}
